// DKF_A_88055419503198
// MI455X (gfx1250) — compile-verified, matrix operand order checked by reading, not run on hardware
//
#include <hip/hip_runtime.h>

typedef __attribute__((ext_vector_type(16))) __bf16 v16bf;
typedef __attribute__((ext_vector_type(8)))  float  v8f;
typedef unsigned short u16;
typedef unsigned int   u32;

#define TT 512
#define BB 256
#define FF 513
#define DXx 256
#define HHH 128
#define ZZz 16

__device__ __forceinline__ u16 f2bf(float x) {
  u32 u = __float_as_uint(x);
  u32 r = u + 0x7FFFu + ((u >> 16) & 1u);
  return (u16)(r >> 16);
}
__device__ __forceinline__ float sigm(float x) { return 1.0f / (1.0f + __expf(-x)); }

__device__ __forceinline__ v8f vzero() {
  v8f a = {0.f, 0.f, 0.f, 0.f, 0.f, 0.f, 0.f, 0.f};
  return a;
}

// 16x32 bf16 fragment load (A layout; also B layout for (N,K)-stored weights).
__device__ __forceinline__ v16bf ldfrag(const u16* tile, int rowStride, int kc) {
  int lane = threadIdx.x & 31;
  const u16* p = tile + (lane & 15) * rowStride + kc + ((lane >> 4) << 3);
  union { v16bf v; uint4 q[2]; } u;
  u.q[0] = *(const uint4*)(p);
  u.q[1] = *(const uint4*)(p + 16);
  return u.v;
}

__device__ __forceinline__ v8f bwmma(v16bf a, v16bf b, v8f c) {
  return __builtin_amdgcn_wmma_f32_16x16x32_bf16(false, a, false, b, (short)0, c, false, false);
}

__device__ __forceinline__ void copy_u16x8(u16* dst, const u16* src, int n) {
  uint4* d = (uint4*)dst;
  const uint4* s = (const uint4*)src;
  int n8 = n >> 3;
  for (int i = threadIdx.x; i < n8; i += blockDim.x) d[i] = s[i];
}

// -------- CDNA5 async global->LDS (16B per lane), tracked by ASYNCcnt -------
__device__ __forceinline__ void async_ld16(unsigned lds_off, unsigned goff,
                                           const void* sbase) {
  asm volatile("global_load_async_to_lds_b128 %0, %1, %2"
               :: "v"(lds_off), "v"(goff), "s"(sbase) : "memory");
}
template <int N>
__device__ __forceinline__ void s_wait_async() {
  asm volatile("s_wait_asynccnt %0" :: "i"(N) : "memory");
}

// ---------------- prep: pack f32 weights -> bf16 (N,K) with zero padding ----
__global__ void k_pack(u16* dst, const float* __restrict__ src,
                       int dstN, int srcN, int dstK, int srcK) {
  int i = blockIdx.x * blockDim.x + threadIdx.x;
  if (i >= dstN * dstK) return;
  int n = i / dstK, k = i - n * dstK;
  float v = (n < srcN && k < srcK) ? src[n * srcK + k] : 0.0f;
  dst[i] = f2bf(v);
}

__global__ void k_addvec(float* dst, const float* a, const float* b, int n) {
  int i = blockIdx.x * blockDim.x + threadIdx.x;
  if (i < n) dst[i] = a[i] + b[i];
}

// ---------------- kernel 1: xg = tanh(xs @ W_xg^T + b_xg), rows r = b*T + t --
__global__ void k_xg(const float* __restrict__ x, const u16* __restrict__ Wxg,
                     const float* __restrict__ bxg, u16* __restrict__ xgbf) {
  extern __shared__ char smem[];
  u16* sW = (u16*)smem;            // 256 x 544 bf16 (whole W_xg, K padded)
  u16* sA = sW + 256 * 544;        // 16 x 544
  const int tid = threadIdx.x;
  const int wave = tid >> 5;
  const int lane = tid & 31;
  const int half = lane >> 4;
  const int nn0  = lane & 15;

  copy_u16x8(sW, Wxg, 256 * 544);
  __syncthreads();

  for (int mt = blockIdx.x; mt < (TT * BB) / 16; mt += gridDim.x) {
    int r0 = mt * 16;
    for (int idx = tid; idx < 16 * 544; idx += 256) {
      int i = idx / 544, f = idx - i * 544;
      int r = r0 + i;
      int b = r >> 9, t = r & 511;
      float v = (f < FF) ? x[(size_t)b * (FF * TT) + (size_t)f * TT + t] : 0.0f;
      sA[idx] = f2bf(v);
    }
    __syncthreads();

    v8f acc0 = vzero(), acc1 = vzero();
    const u16* w0 = sW + (wave * 32) * 544;
    const u16* w1 = sW + (wave * 32 + 16) * 544;
    for (int kc = 0; kc < 544; kc += 32) {
      v16bf af = ldfrag(sA, 544, kc);
      acc0 = bwmma(af, ldfrag(w0, 544, kc), acc0);
      acc1 = bwmma(af, ldfrag(w1, 544, kc), acc1);
    }
    for (int v = 0; v < 8; ++v) {
      int m = v + half * 8;
      size_t row = (size_t)(r0 + m) * DXx;
      int na = wave * 32 + nn0;
      int nb = na + 16;
      xgbf[row + na] = f2bf(tanhf(acc0[v] + bxg[na]));
      xgbf[row + nb] = f2bf(tanhf(acc1[v] + bxg[nb]));
    }
    __syncthreads();
  }
}

// ---------------- kernel 2: gates_x[t,b,:] = xg @ W_ih^T + (b_ih + b_hh) -----
__global__ void k_gatesx(const u16* __restrict__ xgbf, const u16* __restrict__ Wih,
                         const float* __restrict__ bls, float* __restrict__ gatesx) {
  extern __shared__ char smem[];
  u16* sW = (u16*)smem;        // 512 x 256
  u16* sA = sW + 512 * 256;    // 16 x 256
  const int tid = threadIdx.x;
  const int wave = tid >> 5;
  const int lane = tid & 31;
  const int half = lane >> 4;
  const int nn0 = lane & 15;

  copy_u16x8(sW, Wih, 512 * 256);
  __syncthreads();

  for (int mt = blockIdx.x; mt < (TT * BB) / 16; mt += gridDim.x) {
    int r0 = mt * 16;
    copy_u16x8(sA, xgbf + (size_t)r0 * DXx, 16 * DXx);
    __syncthreads();

    v8f acc[4];
    for (int j = 0; j < 4; ++j) acc[j] = vzero();
    for (int kc = 0; kc < DXx; kc += 32) {
      v16bf af = ldfrag(sA, DXx, kc);
      for (int j = 0; j < 4; ++j)
        acc[j] = bwmma(af, ldfrag(sW + (wave * 4 + j) * 16 * DXx, DXx, kc), acc[j]);
    }
    for (int j = 0; j < 4; ++j) {
      int nn = (wave * 4 + j) * 16 + nn0;
      float bv = bls[nn];
      for (int v = 0; v < 8; ++v) {
        int m = v + half * 8;
        int r = r0 + m;
        int b = r >> 9, t = r & 511;
        gatesx[(size_t)t * (BB * 512) + (size_t)b * 512 + nn] = acc[j][v] + bv;
      }
    }
    __syncthreads();
  }
}

// ---------------- kernel 3: reverse LSTM scan, 16 batch rows per WG ----------
// Double-buffered async global->LDS staging of the per-step gates_x tile:
// step t-1's 32KB tile streams in (ASYNCcnt) under step t's WMMAs.
__global__ void k_lstm(const float* __restrict__ gatesx, const u16* __restrict__ Whh,
                       float* __restrict__ gout) {
  extern __shared__ char smem[];
  u16*   sW  = (u16*)smem;                 // 512 x 128 bf16 W_hh
  float* sG0 = (float*)(sW + 512 * 128);   // 16 x 512 gates f32 (buf 0)
  float* sG1 = sG0 + 16 * 512;             // 16 x 512 gates f32 (buf 1)
  float* sC  = sG1 + 16 * 512;             // 16 x 128 c
  u16*   sH  = (u16*)(sC + 16 * 128);      // 16 x 128 h bf16
  const int tid = threadIdx.x;
  const int wave = tid >> 5;
  const int lane = tid & 31;
  const int half = lane >> 4;
  const int nn0 = lane & 15;
  const int b0 = blockIdx.x * 16;

  copy_u16x8(sW, Whh, 512 * 128);
  for (int i = tid; i < 16 * 128; i += 256) { sC[i] = 0.0f; sH[i] = 0; }

  auto issue_tile = [&](int t, float* dstLds) {
    const char* src = (const char*)(gatesx + (size_t)t * (BB * 512) + (size_t)b0 * 512);
    unsigned lbase = (unsigned)(size_t)dstLds;
    for (int i = tid; i < 2048; i += 256)          // 2048 x 16B = 32KB, 8/lane
      async_ld16(lbase + (unsigned)i * 16u, (unsigned)i * 16u, src);
  };
  issue_tile(TT - 1, (TT - 1) & 1 ? sG1 : sG0);

  for (int t = TT - 1; t >= 0; --t) {
    float* sG = (t & 1) ? sG1 : sG0;
    if (t > 0) {
      issue_tile(t - 1, (t & 1) ? sG0 : sG1);      // prefetch next step
      s_wait_async<8>();                           // step-t batch complete
    } else {
      s_wait_async<0>();
    }
    __syncthreads();

    v8f acc[4];
    for (int j = 0; j < 4; ++j) acc[j] = vzero();
    for (int kc = 0; kc < HHH; kc += 32) {
      v16bf af = ldfrag(sH, HHH, kc);
      for (int j = 0; j < 4; ++j)
        acc[j] = bwmma(af, ldfrag(sW + (wave * 4 + j) * 16 * HHH, HHH, kc), acc[j]);
    }
    for (int j = 0; j < 4; ++j) {
      int nn = (wave * 4 + j) * 16 + nn0;
      for (int v = 0; v < 8; ++v) {
        int m = v + half * 8;
        sG[m * 512 + nn] += acc[j][v];
      }
    }
    __syncthreads();

    for (int i = tid; i < 16 * 128; i += 256) {
      int m = i >> 7, hcol = i & 127;
      float ig = sG[m * 512 + hcol];
      float fg = sG[m * 512 + 128 + hcol];
      float gg = sG[m * 512 + 256 + hcol];
      float og = sG[m * 512 + 384 + hcol];
      float c = sigm(fg) * sC[i] + sigm(ig) * tanhf(gg);
      float h = sigm(og) * tanhf(c);
      sC[i] = c;
      sH[i] = f2bf(h);
      gout[(size_t)t * (BB * HHH) + (size_t)(b0 + m) * HHH + hcol] = h;
    }
    __syncthreads();
  }
}

// ---------------- kernel 4: inf_step scan, 16 batch rows per WG --------------
// Async double-buffered staging of the per-step g tile (8KB).
__global__ void k_inf(const float* __restrict__ g, const float* __restrict__ eps,
                      const u16* Wzg0, const u16* Wzg1, const u16* Wim, const u16* Wil,
                      const float* bzg0, const float* bzg1, const float* bim, const float* bil,
                      float* __restrict__ zout) {
  extern __shared__ char smem[];
  u16* sW0 = (u16*)smem;                  // 32 x 32
  u16* sW1 = sW0 + 32 * 32;               // 128 x 32
  u16* sWm = sW1 + 128 * 32;              // 16 x 128
  u16* sWl = sWm + 16 * 128;              // 16 x 128
  u16* sZ  = sWl + 16 * 128;              // 16 x 32 (z bf16, K-padded)
  u16* sH1 = sZ + 16 * 32;                // 16 x 32
  u16* sGt = sH1 + 16 * 32;               // 16 x 128
  float* sGf0 = (float*)(sGt + 16 * 128); // 16 x 128 g tile f32 (buf 0)
  float* sGf1 = sGf0 + 16 * 128;          // 16 x 128 g tile f32 (buf 1)
  float* sZm = sGf1 + 16 * 128;           // 16 x 16
  float* sZl = sZm + 256;                 // 16 x 16
  const int tid = threadIdx.x;
  const int wave = tid >> 5;
  const int lane = tid & 31;
  const int half = lane >> 4;
  const int nn0 = lane & 15;
  const int b0 = blockIdx.x * 16;

  copy_u16x8(sW0, Wzg0, 32 * 32);
  copy_u16x8(sW1, Wzg1, 128 * 32);
  copy_u16x8(sWm, Wim, 16 * 128);
  copy_u16x8(sWl, Wil, 16 * 128);
  for (int i = tid; i < 16 * 32; i += 256) sZ[i] = 0;

  auto issue_g = [&](int t, float* dstLds) {
    const char* src = (const char*)(g + (size_t)t * (BB * HHH) + (size_t)b0 * HHH);
    unsigned lbase = (unsigned)(size_t)dstLds;
    for (int i = tid; i < 512; i += 256)           // 512 x 16B = 8KB, 2/lane
      async_ld16(lbase + (unsigned)i * 16u, (unsigned)i * 16u, src);
  };
  issue_g(0, sGf0);

  for (int t = 0; t < TT; ++t) {
    float* sGf = (t & 1) ? sGf1 : sGf0;
    if (t < TT - 1) {
      issue_g(t + 1, (t & 1) ? sGf0 : sGf1);
      s_wait_async<2>();
    } else {
      s_wait_async<0>();
    }
    __syncthreads();

    if (wave < 2) {                       // hz1 = tanh(z @ Wzg0^T), N=32
      v8f a = vzero();
      v16bf af = ldfrag(sZ, 32, 0);
      a = bwmma(af, ldfrag(sW0 + wave * 16 * 32, 32, 0), a);
      int nn = wave * 16 + nn0;
      float bv = bzg0[nn];
      for (int v = 0; v < 8; ++v) sH1[(v + half * 8) * 32 + nn] = f2bf(tanhf(a[v] + bv));
    }
    __syncthreads();

    {                                     // hz2 + g_t, N=128, one tile per wave
      v8f a = vzero();
      v16bf af = ldfrag(sH1, 32, 0);
      a = bwmma(af, ldfrag(sW1 + wave * 16 * 32, 32, 0), a);
      int nn = wave * 16 + nn0;
      float bv = bzg1[nn];
      for (int v = 0; v < 8; ++v) {
        int m = v + half * 8;
        float gt = 0.5f * (tanhf(a[v] + bv) + sGf[m * 128 + nn]);
        sGt[m * 128 + nn] = f2bf(gt);
      }
    }
    __syncthreads();

    if (wave < 2) {                       // zm (wave0) / zl (wave1), N=16, K=128
      const u16* W = wave ? sWl : sWm;
      v8f a = vzero();
      for (int kc = 0; kc < HHH; kc += 32) {
        v16bf af = ldfrag(sGt, HHH, kc);
        a = bwmma(af, ldfrag(W, HHH, kc), a);
      }
      float* dstm = wave ? sZl : sZm;
      float bv = (wave ? bil : bim)[nn0];
      for (int v = 0; v < 8; ++v) dstm[(v + half * 8) * 16 + nn0] = a[v] + bv;
    }
    __syncthreads();

    {                                     // z = zm + eps * exp(0.5*zl)
      int m = tid >> 4, k = tid & 15;
      float zm = sZm[tid], zl = sZl[tid];
      float e = eps[(size_t)t * (BB * ZZz) + (size_t)(b0 + m) * ZZz + k];
      float zn = zm + e * __expf(0.5f * zl);
      zout[((size_t)t * BB + (b0 + m)) * ZZz + k] = zn;
      sZ[m * 32 + k] = f2bf(zn);
    }
    __syncthreads();
  }
}

// ---------------- kernel 5: fused decoder z -> 128 -> 128 -> exp(@W_gy^T) ----
__global__ void k_dec(const float* __restrict__ z,
                      const u16* Wzx0, const u16* Wzx1, const u16* Wgy,
                      const float* bzx0, const float* bzx1, const float* bgy,
                      float* __restrict__ out) {
  extern __shared__ char smem[];
  u16* sW0 = (u16*)smem;           // 128 x 32
  u16* sW1 = sW0 + 128 * 32;       // 128 x 128
  u16* sWy = sW1 + 128 * 128;      // 528 x 128 (N padded)
  u16* sZ  = sWy + 528 * 128;      // 16 x 32
  u16* sH1 = sZ + 16 * 32;         // 16 x 128
  u16* sH2 = sH1 + 16 * 128;       // 16 x 128
  const int tid = threadIdx.x;
  const int wave = tid >> 5;
  const int lane = tid & 31;
  const int half = lane >> 4;
  const int nn0 = lane & 15;

  copy_u16x8(sW0, Wzx0, 128 * 32);
  copy_u16x8(sW1, Wzx1, 128 * 128);
  copy_u16x8(sWy, Wgy, 528 * 128);
  __syncthreads();

  for (int mt = blockIdx.x; mt < (TT * BB) / 16; mt += gridDim.x) {
    int r0 = mt * 16;                // r = t*B + b
    for (int i = tid; i < 16 * 32; i += 256) {
      int m = i >> 5, k = i & 31;
      sZ[i] = (k < ZZz) ? f2bf(z[(size_t)r0 * ZZz + m * ZZz + k]) : (u16)0;
    }
    __syncthreads();

    {                                 // hy1, N=128
      v8f a = vzero();
      v16bf af = ldfrag(sZ, 32, 0);
      a = bwmma(af, ldfrag(sW0 + wave * 16 * 32, 32, 0), a);
      int nn = wave * 16 + nn0;
      float bv = bzx0[nn];
      for (int v = 0; v < 8; ++v) sH1[(v + half * 8) * 128 + nn] = f2bf(tanhf(a[v] + bv));
    }
    __syncthreads();

    {                                 // hy2, N=128, K=128
      v8f a = vzero();
      for (int kc = 0; kc < 128; kc += 32) {
        v16bf af = ldfrag(sH1, 128, kc);
        a = bwmma(af, ldfrag(sW1 + wave * 16 * 128, 128, kc), a);
      }
      int nn = wave * 16 + nn0;
      float bv = bzx1[nn];
      for (int v = 0; v < 8; ++v) sH2[(v + half * 8) * 128 + nn] = f2bf(tanhf(a[v] + bv));
    }
    __syncthreads();

    for (int nt = wave; nt < 33; nt += 8) {   // y, N=513 padded to 528
      v8f a = vzero();
      for (int kc = 0; kc < 128; kc += 32) {
        v16bf af = ldfrag(sH2, 128, kc);
        a = bwmma(af, ldfrag(sWy + nt * 16 * 128, 128, kc), a);
      }
      int f = nt * 16 + nn0;
      if (f < FF) {
        float bv = bgy[f];
        for (int v = 0; v < 8; ++v) {
          int m = v + half * 8;
          int r = r0 + m;
          int t = r >> 8, b = r & 255;
          out[(size_t)b * (FF * TT) + (size_t)f * TT + t] = __expf(a[v] + bv);
        }
      }
    }
    __syncthreads();
  }
}

// ---------------------------------------------------------------------------
extern "C" void kernel_launch(void* const* d_in, const int* in_sizes, int n_in,
                              void* d_out, int out_size, void* d_ws, size_t ws_size,
                              hipStream_t stream) {
  (void)in_sizes; (void)n_in; (void)out_size; (void)ws_size;
  const float* x    = (const float*)d_in[0];
  const float* eps  = (const float*)d_in[1];
  const float* W_xg = (const float*)d_in[2];
  const float* b_xg = (const float*)d_in[3];
  const float* W_ih = (const float*)d_in[4];
  const float* W_hh = (const float*)d_in[5];
  const float* b_ih = (const float*)d_in[6];
  const float* b_hh = (const float*)d_in[7];
  const float* W_zg0 = (const float*)d_in[8];
  const float* b_zg0 = (const float*)d_in[9];
  const float* W_zg1 = (const float*)d_in[10];
  const float* b_zg1 = (const float*)d_in[11];
  const float* W_im = (const float*)d_in[12];
  const float* b_im = (const float*)d_in[13];
  const float* W_il = (const float*)d_in[14];
  const float* b_il = (const float*)d_in[15];
  const float* W_zx0 = (const float*)d_in[28];
  const float* b_zx0 = (const float*)d_in[29];
  const float* W_zx1 = (const float*)d_in[30];
  const float* b_zx1 = (const float*)d_in[31];
  const float* W_gy = (const float*)d_in[32];
  const float* b_gy = (const float*)d_in[33];
  float* out = (float*)d_out;

  char* ws = (char*)d_ws;
  size_t off = 0;
  auto carve = [&](size_t bytes) {
    char* p = ws + off;
    off += (bytes + 255) & ~(size_t)255;
    return p;
  };
  float* gatesx = (float*)carve((size_t)TT * BB * 512 * 4);
  float* gbuf   = (float*)carve((size_t)TT * BB * HHH * 4);
  float* zbuf   = (float*)carve((size_t)TT * BB * ZZz * 4);
  u16* xgbf  = (u16*)carve((size_t)TT * BB * DXx * 2);
  u16* pWxg  = (u16*)carve(256 * 544 * 2);
  u16* pWih  = (u16*)carve(512 * 256 * 2);
  u16* pWhh  = (u16*)carve(512 * 128 * 2);
  u16* pWzg0 = (u16*)carve(32 * 32 * 2);
  u16* pWzg1 = (u16*)carve(128 * 32 * 2);
  u16* pWim  = (u16*)carve(16 * 128 * 2);
  u16* pWil  = (u16*)carve(16 * 128 * 2);
  u16* pWzx0 = (u16*)carve(128 * 32 * 2);
  u16* pWzx1 = (u16*)carve(128 * 128 * 2);
  u16* pWgy  = (u16*)carve(528 * 128 * 2);
  float* bls = (float*)carve(512 * 4);

  auto pack = [&](u16* dst, const float* src, int dn, int sn, int dk, int sk) {
    int total = dn * dk;
    k_pack<<<(total + 255) / 256, 256, 0, stream>>>(dst, src, dn, sn, dk, sk);
  };
  pack(pWxg,  W_xg,  256, 256, 544, 513);
  pack(pWih,  W_ih,  512, 512, 256, 256);
  pack(pWhh,  W_hh,  512, 512, 128, 128);
  pack(pWzg0, W_zg0,  32,  32,  32,  16);
  pack(pWzg1, W_zg1, 128, 128,  32,  32);
  pack(pWim,  W_im,   16,  16, 128, 128);
  pack(pWil,  W_il,   16,  16, 128, 128);
  pack(pWzx0, W_zx0, 128, 128,  32,  16);
  pack(pWzx1, W_zx1, 128, 128, 128, 128);
  pack(pWgy,  W_gy,  528, 513, 128, 128);
  k_addvec<<<2, 256, 0, stream>>>(bls, b_ih, b_hh, 512);

  size_t smem1 = (size_t)(256 * 544 + 16 * 544) * 2;
  size_t smem2 = (size_t)(512 * 256 + 16 * 256) * 2;
  size_t smem3 = (size_t)512 * 128 * 2 + 2 * 16 * 512 * 4 + 16 * 128 * 4 + 16 * 128 * 2;
  size_t smem4 = (size_t)(32 * 32 + 128 * 32 + 16 * 128 * 2 + 16 * 32 * 2 + 16 * 128) * 2 +
                 (size_t)(2 * 16 * 128 + 256 + 256) * 4;
  size_t smem5 = (size_t)(128 * 32 + 128 * 128 + 528 * 128 + 16 * 32 + 16 * 128 * 2) * 2;

  k_xg    <<<256, 256, smem1, stream>>>(x, pWxg, b_xg, xgbf);
  k_gatesx<<<512, 256, smem2, stream>>>(xgbf, pWih, bls, gatesx);
  k_lstm  <<<16,  256, smem3, stream>>>(gatesx, pWhh, gbuf);
  k_inf   <<<16,  256, smem4, stream>>>(gbuf, eps, pWzg0, pWzg1, pWim, pWil,
                                        b_zg0, b_zg1, b_im, b_il, zbuf);
  k_dec   <<<512, 256, smem5, stream>>>(zbuf, pWzx0, pWzx1, pWgy,
                                        b_zx0, b_zx1, b_gy, out);
}
